// Encoder_936302870755
// MI455X (gfx1250) — matrix operand order repaired and checked by reading, not compiled, not run on hardware
//
#include <hip/hip_runtime.h>

// ---------------- types for WMMA ----------------
typedef __bf16 v16bf __attribute__((ext_vector_type(16)));
typedef __bf16 v8bf  __attribute__((ext_vector_type(8)));
typedef float  v8f   __attribute__((ext_vector_type(8)));

#define DH 256
#define TILE_ROWS 64
#define A_PAD 264   // 264 halves = 528B row stride (multiple of 16B, !=0 mod 64 banks)
#define W_PAD 40    // 40 halves = 80B row stride (multiple of 16B)

// ---------------- threefry2x32 (JAX-compatible, 20 rounds) ----------------
__host__ __device__ inline unsigned tf_rotl(unsigned x, int r) {
  return (x << r) | (x >> (32 - r));
}

__host__ __device__ inline void threefry2x32(unsigned k0, unsigned k1,
                                             unsigned x0, unsigned x1,
                                             unsigned* o0, unsigned* o1) {
  unsigned ks0 = k0, ks1 = k1, ks2 = k0 ^ k1 ^ 0x1BD11BDAu;
  x0 += ks0; x1 += ks1;
#define TF_R4(a,b,c,d) \
  x0 += x1; x1 = tf_rotl(x1,a); x1 ^= x0; \
  x0 += x1; x1 = tf_rotl(x1,b); x1 ^= x0; \
  x0 += x1; x1 = tf_rotl(x1,c); x1 ^= x0; \
  x0 += x1; x1 = tf_rotl(x1,d); x1 ^= x0;
  TF_R4(13,15,26,6)  x0 += ks1; x1 += ks2 + 1u;
  TF_R4(17,29,16,24) x0 += ks2; x1 += ks0 + 2u;
  TF_R4(13,15,26,6)  x0 += ks0; x1 += ks1 + 3u;
  TF_R4(17,29,16,24) x0 += ks1; x1 += ks2 + 4u;
  TF_R4(13,15,26,6)  x0 += ks2; x1 += ks0 + 5u;
#undef TF_R4
  *o0 = x0; *o1 = x1;
}

// ---------------- simple utility kernels ----------------
__global__ __launch_bounds__(256) void zero_kernel(float* __restrict__ p, size_t n) {
  size_t i = (size_t)blockIdx.x * 256 + threadIdx.x;
  if (i < n) p[i] = 0.0f;
}

__global__ __launch_bounds__(256) void degree_kernel(const int* __restrict__ dst,
                                                     float* __restrict__ deg, long E) {
  long e = (long)blockIdx.x * 256 + threadIdx.x;
  if (e < E) atomicAdd(&deg[dst[e]], 1.0f);
}

__global__ __launch_bounds__(256) void rsqrt_kernel(float* __restrict__ dinv, int N) {
  int i = blockIdx.x * 256 + threadIdx.x;
  if (i < N) dinv[i] = rsqrtf(dinv[i] + 1.0f);  // deg over A+I
}

// ---------------- GEMM: C[N x 256] = A[N x 256] @ W[256 x 256], bf16 WMMA, f32 acc ----------------
__global__ __launch_bounds__(256) void gemm_bf16_wmma(const float* __restrict__ A,
                                                      const float* __restrict__ W,
                                                      float* __restrict__ C, int N) {
  __shared__ __align__(16) __bf16 A_s[TILE_ROWS * A_PAD];
  __shared__ __align__(16) __bf16 W_s[DH * W_PAD];

  const int tid  = threadIdx.x;
  const int wv   = tid >> 5;      // 8 waves
  const int lane = tid & 31;
  const int row0 = blockIdx.x * TILE_ROWS;

  // Stage A tile (64 rows x 256) fp32 -> bf16 LDS, coalesced by rows of 256.
  for (int i = tid; i < TILE_ROWS * DH; i += 256) {
    int r = i >> 8, c = i & 255;
    int row = row0 + r;
    float v = (row < N) ? A[(size_t)row * DH + c] : 0.0f;
    A_s[r * A_PAD + c] = (__bf16)v;
  }

  const int rowStrip = (wv & 3) * 16;   // which 16-row strip of the 64-row tile
  const int colHalf  = (wv >> 2) * 128; // which 128-col half of the 256 cols
  const int l16 = lane & 15;
  const bool hi = lane >= 16;

  v8f acc[8] = {};

  for (int k0 = 0; k0 < DH; k0 += 32) {
    __syncthreads();  // previous panel fully consumed
    // Stage W panel W[k0..k0+31][0..255], transposed -> W_s[n][k] (bf16)
    for (int i = tid; i < 32 * DH; i += 256) {
      int k = i >> 8, n = i & 255;
      W_s[n * W_PAD + k] = (__bf16)W[(size_t)(k0 + k) * DH + n];
    }
    __syncthreads();

    // A fragment: lanes 0-15 hold K {0..7,16..23}, lanes 16-31 hold K {8..15,24..31}
    const __bf16* arow = &A_s[(rowStrip + l16) * A_PAD + k0 + (hi ? 8 : 0)];
    v8bf alo = *(const v8bf*)(arow);
    v8bf ahi = *(const v8bf*)(arow + 16);
    v16bf afrag;
#pragma unroll
    for (int i = 0; i < 8; ++i) { afrag[i] = alo[i]; afrag[8 + i] = ahi[i]; }

#pragma unroll
    for (int ct = 0; ct < 8; ++ct) {
      // B fragment: lane holds column n, lanes 0-15 hold K {0..7,16..23}, lanes 16-31 hold K {8..15,24..31}
      const __bf16* brow = &W_s[(colHalf + ct * 16 + l16) * W_PAD + (hi ? 8 : 0)];
      v8bf blo = *(const v8bf*)(brow);
      v8bf bhi = *(const v8bf*)(brow + 16);
      v16bf bfrag;
#pragma unroll
      for (int i = 0; i < 8; ++i) { bfrag[i] = blo[i]; bfrag[8 + i] = bhi[i]; }
      acc[ct] = __builtin_amdgcn_wmma_f32_16x16x32_bf16(
          false, afrag, false, bfrag, (short)0, acc[ct], false, false);
    }
  }

  // C layout: VGPR r -> M = r (lanes 0-15) / M = 8+r (lanes 16-31), N = lane%16
#pragma unroll
  for (int ct = 0; ct < 8; ++ct) {
    int col = colHalf + ct * 16 + l16;
#pragma unroll
    for (int r = 0; r < 8; ++r) {
      int row = row0 + rowStrip + r + (hi ? 8 : 0);
      if (row < N) C[(size_t)row * DH + col] = acc[ct][r];
    }
  }
}

// ---------------- edge scatter: agg[dst] += h[src] * dinv[src]*dinv[dst] ----------------
// 64 threads per edge, float4 payload each -> wide coalesced gather + global_atomic_add_f32.
__global__ __launch_bounds__(256) void scatter_kernel(const float* __restrict__ h,
                                                      const int* __restrict__ src,
                                                      const int* __restrict__ dst,
                                                      const float* __restrict__ dinv,
                                                      float* __restrict__ agg, long E) {
  size_t gid = (size_t)blockIdx.x * 256 + threadIdx.x;
  long e = (long)(gid >> 6);
  if (e >= E) return;
  int c4 = (int)(gid & 63) << 2;
  int s = src[e], d = dst[e];
  float norm = dinv[s] * dinv[d];
  const float4 hv = *(const float4*)&h[(size_t)s * DH + c4];
  float* base = &agg[(size_t)d * DH + c4];
  atomicAdd(base + 0, hv.x * norm);
  atomicAdd(base + 1, hv.y * norm);
  atomicAdd(base + 2, hv.z * norm);
  atomicAdd(base + 3, hv.w * norm);
}

// ---------------- combine: acc = acc + dinv^2 * h + b, optional ReLU + exact JAX dropout ----------------
__global__ __launch_bounds__(256) void combine_kernel(float* __restrict__ acc,
                                                      const float* __restrict__ h,
                                                      const float* __restrict__ dinv,
                                                      const float* __restrict__ bias,
                                                      int N, int applyDrop,
                                                      unsigned key0, unsigned key1) {
  size_t total = (size_t)N * DH;
  size_t i = (size_t)blockIdx.x * 256 + threadIdx.x;
  if (i >= total) return;
  int row = (int)(i >> 8);
  int c = (int)(i & 255);
  float di = dinv[row];
  float v = acc[i] + di * di * h[i] + bias[c];
  if (applyDrop) {
    v = fmaxf(v, 0.0f);  // ReLU
    // JAX threefry counter scheme: iota(n) split in halves -> (i, i+half) pairs
    size_t half = total >> 1;
    unsigned x0, x1; int word;
    if (i < half) { x0 = (unsigned)i; x1 = (unsigned)(i + half); word = 0; }
    else          { x0 = (unsigned)(i - half); x1 = (unsigned)i; word = 1; }
    unsigned o0, o1;
    threefry2x32(key0, key1, x0, x1, &o0, &o1);
    unsigned bits = word ? o1 : o0;
    float u = __uint_as_float((bits >> 9) | 0x3f800000u) - 1.0f;  // uniform [0,1)
    v = (u < 0.8f) ? v * 1.25f : 0.0f;  // keep-prob 0.8, inverted scaling
  }
  acc[i] = v;
}

// ---------------- host launch ----------------
extern "C" void kernel_launch(void* const* d_in, const int* in_sizes, int n_in,
                              void* d_out, int out_size, void* d_ws, size_t ws_size,
                              hipStream_t stream) {
  const float* x  = (const float*)d_in[0];
  const int*   ei = (const int*)d_in[1];
  const float* W1 = (const float*)d_in[2];
  const float* b1 = (const float*)d_in[3];
  const float* W2 = (const float*)d_in[4];
  const float* b2 = (const float*)d_in[5];
  float* out = (float*)d_out;

  const int N = in_sizes[0] / DH;
  const long E = in_sizes[1] / 2;
  const int* src = ei;
  const int* dst = ei + E;

  char* ws = (char*)d_ws;
  size_t dinvBytes = (((size_t)N * 4) + 255) & ~(size_t)255;
  float* dinv = (float*)ws;
  float* h    = (float*)(ws + dinvBytes);          // N*256 f32
  float* agg  = h + (size_t)N * DH;                // N*256 f32

  // folded dropout key: fold_in(key(0), 7) == threefry2x32({0,0},{0,7})
  unsigned fk0, fk1;
  threefry2x32(0u, 0u, 0u, 7u, &fk0, &fk1);

  const size_t tot = (size_t)N * DH;
  const int zbN   = (N + 255) / 256;
  const int zbTot = (int)((tot + 255) / 256);
  const int ebDeg = (int)((E + 255) / 256);
  const int ebSc  = (int)((E * 64 + 255) / 256);
  const int gbGemm = (N + TILE_ROWS - 1) / TILE_ROWS;

  // degrees -> dinv (in-place)
  zero_kernel<<<zbN, 256, 0, stream>>>(dinv, (size_t)N);
  degree_kernel<<<ebDeg, 256, 0, stream>>>(dst, dinv, E);
  rsqrt_kernel<<<zbN, 256, 0, stream>>>(dinv, N);

  // ---- layer 1 ----
  gemm_bf16_wmma<<<gbGemm, 256, 0, stream>>>(x, W1, h, N);
  zero_kernel<<<zbTot, 256, 0, stream>>>(agg, tot);
  scatter_kernel<<<ebSc, 256, 0, stream>>>(h, src, dst, dinv, agg, E);
  combine_kernel<<<zbTot, 256, 0, stream>>>(agg, h, dinv, b1, N, 1, fk0, fk1);
  // agg now holds dropout(relu(layer1))

  // ---- layer 2 ----
  gemm_bf16_wmma<<<gbGemm, 256, 0, stream>>>(agg, W2, h, N);
  zero_kernel<<<zbTot, 256, 0, stream>>>(out, tot);
  scatter_kernel<<<ebSc, 256, 0, stream>>>(h, src, dst, dinv, out, E);
  combine_kernel<<<zbTot, 256, 0, stream>>>(out, h, dinv, b2, N, 0, 0u, 0u);
}
